// ConvolutionalAttention_3015067042131
// MI455X (gfx1250) — compile-verified, matrix operand order checked by reading, not run on hardware
//
#include <hip/hip_runtime.h>
#include <hip/hip_bf16.h>

// ---------------------------------------------------------------------------
// Shapes (fixed by the reference):
//   x:  (16, 128, 64, 64) f32      lk_filter: (64, 64, 13, 13) f32
//   w1: (32, 64) f32  b1: (32,)    w2: (576, 32) f32  b2: (576,)
//   out: (16, 128, 64, 64) f32
// Strategy: 13x13 dense conv as implicit GEMM in bf16 WMMA (f32 accum),
// input slab staged to LDS via the Tensor Data Mover (tensor_load_to_lds),
// fused with the dynamic 3x3 depthwise conv in the epilogue.
// ---------------------------------------------------------------------------

#define B_   16
#define C_   128
#define PD   64      // PDIM
#define H_   64
#define W_   64
#define LK_  13
#define TAPS 169     // 13*13

typedef __attribute__((ext_vector_type(16))) __bf16        v16bf;
typedef __attribute__((ext_vector_type(8)))  float         v8f;
typedef __attribute__((ext_vector_type(4)))  unsigned int  v4u;
typedef __attribute__((ext_vector_type(8)))  int           v8i_;
typedef __attribute__((ext_vector_type(4)))  int           v4i_;

static __device__ __forceinline__ unsigned short f2bf(float f) {
    unsigned int u = __float_as_uint(f);
    u += 0x7FFFu + ((u >> 16) & 1u);           // round-to-nearest-even
    return (unsigned short)(u >> 16);
}
static __device__ __forceinline__ float bf2f(unsigned short s) {
    return __uint_as_float(((unsigned int)s) << 16);
}

// ---------------------------------------------------------------------------
// Pack x[:, :64] (NCHW f32) -> xT[b][h][w][c] bf16 (channel-last)
// ---------------------------------------------------------------------------
__global__ __launch_bounds__(256) void k_pack_x(const float* __restrict__ x,
                                                unsigned short* __restrict__ xT) {
    int t = blockIdx.x * 256 + threadIdx.x;          // over 16*64*64*64
    if (t >= B_ * PD * H_ * W_) return;
    int ww = t & 63, hh = (t >> 6) & 63, c = (t >> 12) & 63, b = t >> 18;
    float v = x[((b * C_ + c) << 12) + (hh << 6) + ww];
    xT[(((b * H_ + hh) * W_ + ww) * PD) + c] = f2bf(v);
}

// ---------------------------------------------------------------------------
// Pack lk_filter (Co,Ci,13,13 f32) into WMMA B-fragment order:
//   Fpack[((tap*2 + kt)*4 + nt)*512 + lane*16 + half]   (bf16)
// B is 32(K=ci)x16(N=co); lane/half mapping mirrors the 16-bit A layout.
// ---------------------------------------------------------------------------
__global__ __launch_bounds__(256) void k_pack_f(const float* __restrict__ f,
                                                unsigned short* __restrict__ Fp) {
    int t = blockIdx.x * 256 + threadIdx.x;          // over 64*64*169
    if (t >= PD * PD * TAPS) return;
    int tap = t % TAPS;
    int ci  = (t / TAPS) % PD;
    int co  = t / (TAPS * PD);
    float v = f[(co * PD + ci) * TAPS + tap];

    int kt = ci >> 5, kk = ci & 31;
    int nt = co >> 4, n  = co & 15;
    int hh   = kk & 1;
    int kmid = (kk >> 3) & 1;
    int khi  = kk >> 4;
    int vr   = khi * 4 + ((kk & 7) >> 1);
    int lane = n + 16 * kmid;
    int half = vr * 2 + hh;
    Fp[(((tap * 2 + kt) * 4 + nt) * 32 + lane) * 16 + half] = f2bf(v);
}

// ---------------------------------------------------------------------------
// pooled[b,c] = mean over HxW of x[b,c].  One block per (b,c).
// ---------------------------------------------------------------------------
__global__ __launch_bounds__(256) void k_pool(const float* __restrict__ x,
                                              float* __restrict__ pooled) {
    __shared__ float red[256];
    int bc = blockIdx.x;                 // 0..1023
    int b = bc >> 6, c = bc & 63;
    const float* p = x + ((b * C_ + c) << 12);
    float s = 0.f;
    for (int i = threadIdx.x; i < 4096; i += 256) s += p[i];
    red[threadIdx.x] = s;
    __syncthreads();
    for (int st = 128; st > 0; st >>= 1) {
        if (threadIdx.x < st) red[threadIdx.x] += red[threadIdx.x + st];
        __syncthreads();
    }
    if (threadIdx.x == 0) pooled[bc] = red[0] * (1.0f / 4096.0f);
}

// ---------------------------------------------------------------------------
// MLP: h = gelu(pooled @ w1^T + b1);  dyn = h @ w2^T + b2.  One block per b.
// ---------------------------------------------------------------------------
__global__ __launch_bounds__(128) void k_mlp(const float* __restrict__ pooled,
                                             const float* __restrict__ w1,
                                             const float* __restrict__ b1,
                                             const float* __restrict__ w2,
                                             const float* __restrict__ b2,
                                             float* __restrict__ dyn) {
    __shared__ float sp[64];
    __shared__ float sh[32];
    int b = blockIdx.x, tid = threadIdx.x;
    if (tid < 64) sp[tid] = pooled[b * 64 + tid];
    __syncthreads();
    if (tid < 32) {
        float a = b1[tid];
        for (int c = 0; c < 64; ++c) a += sp[c] * w1[tid * 64 + c];
        sh[tid] = 0.5f * a * (1.0f + erff(a * 0.70710678118f));  // exact gelu
    }
    __syncthreads();
    for (int i = tid; i < 576; i += 128) {
        float a = b2[i];
        for (int j = 0; j < 32; ++j) a += sh[j] * w2[i * 32 + j];
        dyn[b * 576 + i] = a;
    }
}

// ---------------------------------------------------------------------------
// Main conv kernel: one workgroup per (b, h) output row; 4 waves of 32.
// Wave w owns pixels [16w, 16w+16) x all 64 output channels.
// Input slab (valid rows x 64 cols x 32 ch, bf16) staged via TDM.
// Implicit GEMM over 169 taps x 2 channel-halves, bf16 WMMA, f32 accum.
// OOB rows/cols handled with one per-lane predicate + cndmask-zeroed A
// fragment, keeping the inner loop branch-free (flat CFG -> clean WMMA
// chaining without accumulator copies / hazard NOPs).
// ---------------------------------------------------------------------------
__global__ __launch_bounds__(128) void k_conv(const unsigned short* __restrict__ xT,
                                              const unsigned short* __restrict__ Fp,
                                              const float* __restrict__ dyn,
                                              float* __restrict__ out) {
    // LDS slab: [row][col 0..63][ch 0..31] bf16 = 13*64*32*2 = 53248 B
    __shared__ unsigned int tile32[13 * 64 * 16];

    const int b = blockIdx.x >> 6;
    const int h = blockIdx.x & 63;
    const int lane  = threadIdx.x & 31;
    const int wave  = threadIdx.x >> 5;     // 0..3 -> M-tile
    const int mtile = wave;

    const int hh0   = (h - 6 < 0) ? 0 : h - 6;          // first valid input row
    const int hh1   = (h + 6 > 63) ? 63 : h + 6;        // last valid input row
    const int nrows = hh1 - hh0 + 1;

    v8f acc[4] = {};                        // 4 N-tiles of f32 16x16

    const int m    = lane & 15;
    const int kmid = lane >> 4;
    const int wpix = mtile * 16 + m;        // this lane's A-row pixel

    for (int chalf = 0; chalf < 2; ++chalf) {
        // ---- TDM: DMA the slab (nrows x 64 cols x 32 ch of bf16) into LDS.
        // 3-D tile: X = 32 ch (contiguous), Y = 64 cols (stride 64 elem),
        //           Z = nrows     (stride 64*64 elem). data_size = 2 bytes.
        if (threadIdx.x < 32) {
            unsigned long long ga = (unsigned long long)(uintptr_t)xT +
                2ull * ((unsigned long long)(b * H_ + hh0) * 4096ull +
                        (unsigned long long)(chalf * 32));
            unsigned int ldsoff = (unsigned int)(uintptr_t)(void*)tile32;

            v4u g0 = { 1u,                                  // count=1, user mode
                       ldsoff,                              // lds_addr
                       (unsigned int)ga,                    // global_addr[31:0]
                       (unsigned int)((ga >> 32) & 0x01FFFFFFu) | 0x80000000u };
                                                            // addr[56:32] | type=2
            v8i_ g1 = { (int)(1u << 16),                    // data_size = 2B
                        (int)(64u << 16),                   // tensor_dim0 = 64 (lo16)
                        (int)(64u << 16),                   // dim0 hi=0 | tensor_dim1=64 lo16
                        (int)(32u << 16),                   // dim1 hi=0 | tile_dim0 = 32
                        (int)(64u | ((unsigned)nrows << 16)), // tile_dim1=64 | tile_dim2=nrows
                        (int)64,                            // tensor_dim0_stride = 64
                        (int)(4096u << 16),                 // stride0 hi=0 | tensor_dim1_stride lo16
                        0 };                                // tensor_dim1_stride hi = 0
            v4i_ g2 = { 64, 0, 0, 0 };                      // tensor_dim2=64; tile_dim3=0
            v4i_ g3 = { 0, 0, 0, 0 };
            v8i_ g4 = { 0, 0, 0, 0, 0, 0, 0, 0 };          // extra group (clang-23 form)
            __builtin_amdgcn_tensor_load_to_lds(g0, g1, g2, g3, g4, 0);
            __builtin_amdgcn_s_wait_tensorcnt(0);
        }
        __syncthreads();

        // ---- flat, branch-free tap loop (169 iterations) ------------------
        for (int tap = 0; tap < TAPS; ++tap) {
            const int ky = tap / 13;
            const int kx = tap - ky * 13;

            const int hh  = h + ky - 6;
            const bool rok = (unsigned)hh < 64u;
            const int rr  = rok ? (hh - hh0) : 0;        // clamped slab row

            const int col  = wpix + kx - 6;
            const bool cok = (unsigned)col < 64u;
            const int colc = cok ? col : 0;              // clamped slab col
            const bool ok  = rok && cok;                 // single lane predicate

            // A fragment: 16(pix) x 32(ch) per ISA 16-bit A layout
            union { v16bf v; unsigned int u[8]; } au;
            const int rowbase = (rr * 64 + colc) * 16;
#pragma unroll
            for (int vr = 0; vr < 8; ++vr) {
                int kb = (vr < 4 ? 2 * vr : 16 + 2 * (vr - 4)) + 8 * kmid;
                unsigned int val = tile32[rowbase + (kb >> 1)];
                au.u[vr] = ok ? val : 0u;
            }

            const unsigned short* fbase =
                Fp + (((tap * 2 + chalf) * 4) * 32 + lane) * 16;
            if (tap + 1 < TAPS) {   // prefetch next tap's B fragments
                __builtin_prefetch(
                    Fp + (((tap + 1) * 2 + chalf) * 4 * 32 + lane) * 16, 0, 1);
            }
#pragma unroll
            for (int nt = 0; nt < 4; ++nt) {
                v16bf bf = *(const v16bf*)(fbase + nt * 512);
                acc[nt] = __builtin_amdgcn_wmma_f32_16x16x32_bf16(
                    false, au.v, false, bf, (short)0, acc[nt], false, false);
            }
        }
        __syncthreads();
    }

    // ---- epilogue: fuse dynamic 3x3 depthwise conv, store f32 -------------
    const int n  = lane & 15;
    const int hi = lane >> 4;
#pragma unroll
    for (int nt = 0; nt < 4; ++nt) {
        const int c = nt * 16 + n;
        float cf[9];
#pragma unroll
        for (int t = 0; t < 9; ++t) cf[t] = dyn[(b * 64 + c) * 9 + t];
#pragma unroll
        for (int r = 0; r < 8; ++r) {
            const int w = mtile * 16 + r + 8 * hi;   // pixel (C/D layout)
            float s = 0.f;
#pragma unroll
            for (int dy = -1; dy <= 1; ++dy) {
#pragma unroll
                for (int dx = -1; dx <= 1; ++dx) {
                    int hh = h + dy, ww = w + dx;
                    float xv = 0.f;
                    if ((unsigned)hh < 64u && (unsigned)ww < 64u)
                        xv = bf2f(xT[(((b * H_ + hh) * W_ + ww) * PD) + c]);
                    s += xv * cf[(dy + 1) * 3 + (dx + 1)];
                }
            }
            out[((b * C_ + c) << 12) + (h << 6) + w] = acc[nt][r] + s;
        }
    }
}

// ---------------------------------------------------------------------------
// Pass-through copy of x[:, 64:] -> out[:, 64:]  (float4 vectorized)
// ---------------------------------------------------------------------------
__global__ __launch_bounds__(256) void k_copy(const float* __restrict__ x,
                                              float* __restrict__ out) {
    int t = blockIdx.x * 256 + threadIdx.x;     // over (16*64*4096)/4 float4
    if (t >= (B_ * PD * 4096) / 4) return;
    int b  = t / 65536;
    int o4 = t - b * 65536;
    const float4* src = (const float4*)x;
    float4*       dst = (float4*)out;
    int idx = b * (C_ * 1024) + 64 * 1024 + o4;
    dst[idx] = src[idx];
}

// ---------------------------------------------------------------------------
extern "C" void kernel_launch(void* const* d_in, const int* in_sizes, int n_in,
                              void* d_out, int out_size, void* d_ws, size_t ws_size,
                              hipStream_t stream) {
    const float* x  = (const float*)d_in[0];
    const float* lk = (const float*)d_in[1];
    const float* w1 = (const float*)d_in[2];
    const float* b1 = (const float*)d_in[3];
    const float* w2 = (const float*)d_in[4];
    const float* b2 = (const float*)d_in[5];
    float* out = (float*)d_out;

    char* ws = (char*)d_ws;
    unsigned short* xT = (unsigned short*)ws;                       // 8,388,608 B
    unsigned short* Fp = (unsigned short*)(ws + 8388608);           // 1,384,448 B
    float* pooled      = (float*)(ws + 8388608 + 1384448);          //     4,096 B
    float* dynp        = (float*)(ws + 8388608 + 1384448 + 4096);   //    36,864 B

    k_pack_x<<<(B_ * PD * H_ * W_ + 255) / 256, 256, 0, stream>>>(x, xT);
    k_pack_f<<<(PD * PD * TAPS + 255) / 256, 256, 0, stream>>>(lk, Fp);
    k_pool  <<<B_ * PD, 256, 0, stream>>>(x, pooled);
    k_mlp   <<<B_, 128, 0, stream>>>(pooled, w1, b1, w2, b2, dynp);
    k_conv  <<<B_ * H_, 128, 0, stream>>>(xT, Fp, dynp, out);
    k_copy  <<<(B_ * PD * 4096 / 4 + 255) / 256, 256, 0, stream>>>(x, out);
}
